// ModifiedNeighborhoodAttention_575525618126
// MI455X (gfx1250) — matrix operand order repaired and checked by reading, not compiled, not run on hardware
//
#include <hip/hip_runtime.h>

typedef __attribute__((ext_vector_type(16))) _Float16 v16h;
typedef __attribute__((ext_vector_type(8)))  _Float16 v8h;
typedef __attribute__((ext_vector_type(8)))  float    v8f;

#define DIMC 128
#define NHD  4
#define HDC  32
#define HSZ  64
#define WSZ  64
#define BB   2
#define KSZ  7
#define DILC 2
#define MROWS (BB*HSZ*WSZ)          // 8192
#define SCALE_Q 0.17677669529663687f // 32^-0.5

// ---------------------------------------------------------------------------
// Stage: convert four f32 [k][n] weight matrices -> f16 transposed [n][k]
// ---------------------------------------------------------------------------
__global__ __launch_bounds__(256) void stage_weights(
    const float* __restrict__ Wq, const float* __restrict__ Wk,
    const float* __restrict__ Wv, const float* __restrict__ Wo,
    _Float16* __restrict__ Wt) {
  int i = blockIdx.x * 256 + threadIdx.x;      // 0 .. 4*16384-1
  int m = i >> 14;
  int e = i & 16383;
  int n = e & 127;
  int k = e >> 7;
  const float* W = (m == 0) ? Wq : (m == 1) ? Wk : (m == 2) ? Wv : Wo;
  Wt[(size_t)m * 16384 + (size_t)n * 128 + k] = (_Float16)W[(size_t)k * 128 + n];
}

// ---------------------------------------------------------------------------
// WMMA GEMM: out[M=8192][128] = (A[8192][128] @ W[128][128] + bias) * scale
// Wt is f16 transposed [n][k]. One 16-row tile per block; 8 waves = 8 N-tiles.
// ---------------------------------------------------------------------------
__global__ __launch_bounds__(256) void gemm_bias(
    const float* __restrict__ A,
    const _Float16* __restrict__ Wt,
    const float* __restrict__ bias,
    float* __restrict__ out,
    float scale) {
  __shared__ __align__(16) _Float16 Alds[16 * 128];
  const int tid = threadIdx.x;
  const int rowbase = blockIdx.x * 16;

  // Cooperative stage of the 16x128 A tile into LDS as f16 (8 halves/thread).
  {
    int e = tid * 8;                 // 2048 halves total
    const float* src = A + (size_t)rowbase * 128 + e;
    float4 f0 = *(const float4*)(src);
    float4 f1 = *(const float4*)(src + 4);
    v8h h;
    h[0] = (_Float16)f0.x; h[1] = (_Float16)f0.y;
    h[2] = (_Float16)f0.z; h[3] = (_Float16)f0.w;
    h[4] = (_Float16)f1.x; h[5] = (_Float16)f1.y;
    h[6] = (_Float16)f1.z; h[7] = (_Float16)f1.w;
    *(v8h*)&Alds[e] = h;
  }
  __syncthreads();

  const int wave = tid >> 5;         // N-tile 0..7
  const int lane = tid & 31;
  const int hl   = lane >> 4;        // half-wave
  const int lrow = lane & 15;
  const int ncol = wave * 16 + lrow;
  const _Float16* wcol = Wt + (size_t)ncol * 128;

  v8f acc = {};
#pragma unroll
  for (int kt = 0; kt < 4; ++kt) {
    const int kb = kt * 32;
    // A fragment (16x32 f16): a[0..7] = A[m][kb+8*hl .. +7], a[8..15] = +16
    v8h a0 = *(const v8h*)&Alds[lrow * 128 + kb + 8 * hl];
    v8h a1 = *(const v8h*)&Alds[lrow * 128 + kb + 16 + 8 * hl];
    // B fragment (32x16 f16): b[0..7] = Wt[ncol][kb+8*hl .. +7], b[8..15] = +16
    v8h b0 = *(const v8h*)&wcol[kb + 8 * hl];
    v8h b1 = *(const v8h*)&wcol[kb + 16 + 8 * hl];
    v16h a, b;
#pragma unroll
    for (int t = 0; t < 8; ++t) {
      a[t] = a0[t]; a[8 + t] = a1[t];
      b[t] = b0[t]; b[8 + t] = b1[t];
    }
    acc = __builtin_amdgcn_wmma_f32_16x16x32_f16(
        false, a, false, b, (short)0, acc, false, false);
  }

  const float bn = bias[ncol];
  const int r0 = rowbase + 8 * hl;   // D: VGPR r -> M = r + 8*hl, N = ncol
#pragma unroll
  for (int r = 0; r < 8; ++r) {
    out[(size_t)(r0 + r) * 128 + ncol] = (acc[r] + bn) * scale;
  }
}

// ---------------------------------------------------------------------------
// NATTEN 2D dilated neighborhood attention, online softmax, 1 thread/pixel-head
// ---------------------------------------------------------------------------
__device__ __forceinline__ void natten_idx(int i, int& s, int& p) {
  const int ns = KSZ / 2;                        // 3
  if (i - ns * DILC < 0) {
    s = i % DILC;
    p = KSZ - 1 - (i / DILC);
  } else if (i + ns * DILC >= HSZ) {             // H == W == 64
    int imodd = i % DILC;
    int a = (HSZ / DILC) * DILC;
    int b = HSZ - a;
    if (imodd < b) s = HSZ - b + imodd - 2 * ns * DILC;
    else           s = a + imodd - KSZ * DILC;
    p = (HSZ - i - 1) / DILC;
  } else {
    s = i - ns * DILC;
    p = ns;
  }
}

__global__ __launch_bounds__(256) void natten2d(
    const float* __restrict__ qh, const float* __restrict__ kh,
    const float* __restrict__ vh, const float* __restrict__ rpb,
    float* __restrict__ outb) {
  int idx = blockIdx.x * 256 + threadIdx.x;      // (b, nh, h, w)
  int w  = idx & 63;
  int h  = (idx >> 6) & 63;
  int nh = (idx >> 12) & 3;
  int b  = idx >> 14;

  int sh, ph, sw, pw;
  natten_idx(h, sh, ph);
  natten_idx(w, sw, pw);

  const size_t pix = ((size_t)(b * HSZ + h) * WSZ + w) * DIMC + nh * HDC;
  float4 q[8];
  const float4* qp = (const float4*)(qh + pix);
#pragma unroll
  for (int t = 0; t < 8; ++t) q[t] = qp[t];

  float mrun = -1e30f, lrun = 0.f;
  float acc[32];
#pragma unroll
  for (int c = 0; c < 32; ++c) acc[c] = 0.f;

  const float* rp = rpb + ((size_t)nh * 13 + ph) * 13 + pw;

  for (int ki = 0; ki < KSZ; ++ki) {
    int hk = sh + DILC * ki;
    for (int kj = 0; kj < KSZ; ++kj) {
      int wk = sw + DILC * kj;
      size_t np = ((size_t)(b * HSZ + hk) * WSZ + wk) * DIMC + nh * HDC;
      const float4* kp = (const float4*)(kh + np);
      float dot = 0.f;
#pragma unroll
      for (int t = 0; t < 8; ++t) {
        float4 kv = kp[t];
        dot = fmaf(q[t].x, kv.x, dot);
        dot = fmaf(q[t].y, kv.y, dot);
        dot = fmaf(q[t].z, kv.z, dot);
        dot = fmaf(q[t].w, kv.w, dot);
      }
      float logit = dot + rp[ki * 13 + kj];
      float mn    = fmaxf(mrun, logit);
      float alpha = __expf(mrun - mn);
      float p     = __expf(logit - mn);
      lrun = lrun * alpha + p;
      const float4* vp = (const float4*)(vh + np);
#pragma unroll
      for (int t = 0; t < 8; ++t) {
        float4 vv = vp[t];
        acc[4 * t + 0] = fmaf(p, vv.x, acc[4 * t + 0] * alpha);
        acc[4 * t + 1] = fmaf(p, vv.y, acc[4 * t + 1] * alpha);
        acc[4 * t + 2] = fmaf(p, vv.z, acc[4 * t + 2] * alpha);
        acc[4 * t + 3] = fmaf(p, vv.w, acc[4 * t + 3] * alpha);
      }
      mrun = mn;
    }
  }

  float inv = 1.f / lrun;
  float* op = outb + pix;
#pragma unroll
  for (int t = 0; t < 8; ++t) {
    float4 o;
    o.x = acc[4 * t + 0] * inv;
    o.y = acc[4 * t + 1] * inv;
    o.z = acc[4 * t + 2] * inv;
    o.w = acc[4 * t + 3] * inv;
    *(float4*)(op + 4 * t) = o;
  }
}

// ---------------------------------------------------------------------------
extern "C" void kernel_launch(void* const* d_in, const int* in_sizes, int n_in,
                              void* d_out, int out_size, void* d_ws, size_t ws_size,
                              hipStream_t stream) {
  (void)in_sizes; (void)n_in; (void)out_size; (void)ws_size;
  const float* q   = (const float*)d_in[0];
  const float* k   = (const float*)d_in[1];
  const float* v   = (const float*)d_in[2];
  const float* Wq  = (const float*)d_in[3];
  const float* bq  = (const float*)d_in[4];
  const float* Wk  = (const float*)d_in[5];
  const float* bk  = (const float*)d_in[6];
  const float* Wv  = (const float*)d_in[7];
  const float* bv  = (const float*)d_in[8];
  const float* rpb = (const float*)d_in[9];
  const float* Wo  = (const float*)d_in[10];
  const float* bo  = (const float*)d_in[11];

  const size_t NELT = (size_t)MROWS * DIMC;     // 1048576 floats per tensor
  float* qh  = (float*)d_ws;
  float* khb = qh  + NELT;
  float* vhb = khb + NELT;
  float* at  = vhb + NELT;
  _Float16* Wt = (_Float16*)(at + NELT);        // 4 * 16384 halves

  stage_weights<<<256, 256, 0, stream>>>(Wq, Wk, Wv, Wo, Wt);

  gemm_bias<<<MROWS / 16, 256, 0, stream>>>(q, Wt,          bq, qh,  SCALE_Q);
  gemm_bias<<<MROWS / 16, 256, 0, stream>>>(k, Wt + 16384,  bk, khb, 1.0f);
  gemm_bias<<<MROWS / 16, 256, 0, stream>>>(v, Wt + 32768,  bv, vhb, 1.0f);

  natten2d<<<(BB * NHD * HSZ * WSZ) / 256, 256, 0, stream>>>(qh, khb, vhb, rpb, at);

  gemm_bias<<<MROWS / 16, 256, 0, stream>>>(at, Wt + 49152, bo, (float*)d_out, 1.0f);
}
